// CPMoudle_25185688224387
// MI455X (gfx1250) — matrix operand order repaired and checked by reading, not compiled, not run on hardware
//
#include <hip/hip_runtime.h>
#include <cstdint>
#include <cstddef>

// ---------------------------------------------------------------------------
// Types / WMMA helpers (CDNA5 / gfx1250, wave32)
// ---------------------------------------------------------------------------
typedef __attribute__((ext_vector_type(16))) __bf16 bf16x16;
typedef __attribute__((ext_vector_type(8)))  __bf16 bf16x8;
typedef __attribute__((ext_vector_type(8)))  float  f32x8;

#define DEVI __device__ __forceinline__

static constexpr int BATCH = 4;
static constexpr int NPIX  = 4096;   // 64*64
static constexpr int CINF  = 2048;   // feat channels
static constexpr int CR    = 256;    // reduced channels
static constexpr int CTOT  = 2560;   // concat channels (feat + intra + inter)

DEVI f32x8 wmma_bf16(bf16x16 a, bf16x16 b, f32x8 c) {
  // D = A(16x32) * B(32x16) + C, fp32 accumulate
  return __builtin_amdgcn_wmma_f32_16x16x32_bf16(false, a, false, b,
                                                 (short)0, c, false, false);
}
DEVI f32x8 zero8() { f32x8 z = {0.f,0.f,0.f,0.f,0.f,0.f,0.f,0.f}; return z; }
DEVI bf16x16 cat8(bf16x8 lo, bf16x8 hi) {
  return __builtin_shufflevector(lo, hi, 0,1,2,3,4,5,6,7,8,9,10,11,12,13,14,15);
}

// ---------------------------------------------------------------------------
// Weight conversion: OIHW fp32 -> fragment-ready bf16 A-blobs.
// A-tile (16 M x 32 K) blob = 32 lanes * 16 elems contiguous.
//   lanes 0-15 : m = lane,     slots 0-15 -> k 0-15
//   lanes 16-31: m = lane-16,  slots 0-15 -> k 16-31
// Blob index: ((tap*KT + kt)*16 + mt)*512 + lane*16 + slot
// ---------------------------------------------------------------------------
__global__ void k_wfrag(const float* __restrict__ w, __bf16* __restrict__ dst,
                        int I, int taps, int total) {
  int idx = blockIdx.x * 256 + threadIdx.x;
  if (idx >= total) return;
  int tap = idx % taps;
  int rem = idx / taps;
  int i = rem % I;
  int o = rem / I;
  float v = w[idx];                    // flat OIHW == ((o*I+i)*taps+tap)
  int mt = o >> 4, m = o & 15;
  int kt = i >> 5, kl = i & 31;
  int half = (kl >> 4) & 1;
  int lane = m + 16 * half;
  int slot = kl & 15;
  size_t KT = (size_t)(I >> 5);
  size_t d = (((size_t)tap * KT + kt) * 16 + mt) * 512 + lane * 16 + slot;
  dst[d] = (__bf16)v;
}

// Zero page used for implicit conv padding (CTOT bf16 elements)
__global__ void k_zero(__bf16* __restrict__ z, int n) {
  int i = blockIdx.x * 256 + threadIdx.x;
  if (i < n) z[i] = (__bf16)0.0f;
}

// feat NCHW fp32 -> NHWC bf16 via LDS 32x32 tile transpose (coalesced both ways)
__global__ __launch_bounds__(256) void k_tonhwc(const float* __restrict__ feat,
                                                __bf16* __restrict__ out) {
  __shared__ float t[32][33];
  int blk = blockIdx.x;                 // b * 64 * 128 blocks
  int pblk = blk & 127;                 // 4096/32
  int cblk = (blk >> 7) & 63;           // 2048/32
  int b = blk >> 13;
  int p0 = pblk * 32, c0 = cblk * 32;
  int tp = threadIdx.x & 31, tc = threadIdx.x >> 5;   // tc: 0..7
#pragma unroll
  for (int r = 0; r < 4; ++r) {
    int c = tc + r * 8;
    t[c][tp] = feat[((size_t)(b * CINF + c0 + c)) * NPIX + p0 + tp];
  }
  __syncthreads();
#pragma unroll
  for (int r = 0; r < 4; ++r) {
    int p = tc + r * 8;
    out[((size_t)(b * NPIX + p0 + p)) * CTOT + c0 + tp] = (__bf16)t[tp][p];
  }
}

// Fold conv-bias + inference BN into per-channel scale/offset
__global__ void k_bnprep(const float* __restrict__ g, const float* __restrict__ b,
                         const float* __restrict__ m, const float* __restrict__ v,
                         const float* __restrict__ cb,
                         float* __restrict__ A, float* __restrict__ Bo) {
  int c = threadIdx.x;
  float s = g[c] * rsqrtf(v[c] + 1e-5f);
  float bias = cb ? cb[c] : 0.0f;
  A[c] = s;
  Bo[c] = (bias - m[c]) * s + b[c];
}

// ---------------------------------------------------------------------------
// Implicit-GEMM 3x3 conv (pad 1), NHWC bf16 in -> NCHW fp32 out, BN+ReLU.
// Block 128 = 4 waves; workgroup tile 64 couts x 64 pixels (one image row).
// Wave tile: 32 couts x 32 pixels -> 4 accumulators, 4 loads : 4 WMMA per
// k-step. Padding lanes read a zero page (no EXEC divergence in the loop).
// ---------------------------------------------------------------------------
__global__ __launch_bounds__(128) void k_conv3x3(
    const __bf16* __restrict__ inNHWC, int Cin,
    const __bf16* __restrict__ wfrag,
    const __bf16* __restrict__ zpad,
    const float* __restrict__ bnA, const float* __restrict__ bnB,
    float* __restrict__ outNCHW) {
  const int lane = threadIdx.x & 31;
  const int wave = threadIdx.x >> 5;
  const int h = lane >> 4, ncol = lane & 15;
  const int m0 = blockIdx.x * 64 + (wave >> 1) * 32;
  const int n0 = blockIdx.y * 64 + (wave & 1) * 32;
  const int b = n0 >> 12, p0 = n0 & (NPIX - 1);
  const int y = p0 >> 6, x0 = p0 & 63;
  const int KT = Cin >> 5;
  const __bf16* zp = zpad + h * 16;
  f32x8 acc00 = zero8(), acc01 = zero8(), acc10 = zero8(), acc11 = zero8();

  for (int tap = 0; tap < 9; ++tap) {
    int dy = tap / 3 - 1, dx = tap % 3 - 1;
    int iy = y + dy;
    int ix0 = x0 + ncol + dx;
    int ix1 = ix0 + 16;
    bool okY = (unsigned)iy < 64u;
    const __bf16* src0 = (okY && (unsigned)ix0 < 64u)
        ? inNHWC + ((size_t)(b * NPIX + iy * 64 + ix0)) * CTOT + h * 16 : zp;
    const __bf16* src1 = (okY && (unsigned)ix1 < 64u)
        ? inNHWC + ((size_t)(b * NPIX + iy * 64 + ix1)) * CTOT + h * 16 : zp;
    const __bf16* wb0 =
        wfrag + (((size_t)tap * KT) * 16 + (m0 >> 4)) * 512 + lane * 16;
    for (int kt = 0; kt < KT; ++kt) {
      const __bf16* wb = wb0 + (size_t)kt * (16 * 512);
      bf16x16 a0 = *(const bf16x16*)wb;
      bf16x16 a1 = *(const bf16x16*)(wb + 512);
      bf16x16 b0 = *(const bf16x16*)(src0 + (size_t)kt * 32);
      bf16x16 b1 = *(const bf16x16*)(src1 + (size_t)kt * 32);
      acc00 = wmma_bf16(a0, b0, acc00);
      acc10 = wmma_bf16(a1, b0, acc10);
      acc01 = wmma_bf16(a0, b1, acc01);
      acc11 = wmma_bf16(a1, b1, acc11);
    }
  }
#pragma unroll
  for (int t = 0; t < 2; ++t) {
#pragma unroll
    for (int u = 0; u < 2; ++u) {
      f32x8 a = t ? (u ? acc11 : acc10) : (u ? acc01 : acc00);
      int mb = m0 + t * 16 + 8 * h;
      int pp = p0 + u * 16 + ncol;
#pragma unroll
      for (int r = 0; r < 8; ++r) {
        int m = mb + r;
        float v = a[r] * bnA[m] + bnB[m];
        v = v > 0.f ? v : 0.f;
        outNCHW[((size_t)(b * CR + m)) * NPIX + pp] = v;
      }
    }
  }
}

// Depthwise 1x11 / 11x1 conv, fp32, NCHW [B,256,64,64]
__global__ void k_dw(const float* __restrict__ in, const float* __restrict__ w,
                     const float* __restrict__ bias, float* __restrict__ out,
                     int vertical, int accumulate) {
  int idx = blockIdx.x * 256 + threadIdx.x;     // B*256*4096
  if (idx >= BATCH * CR * NPIX) return;
  int p = idx & (NPIX - 1);
  int c = (idx >> 12) & (CR - 1);
  int y = p >> 6, x = p & 63;
  const float* base = in + (size_t)(idx - p);
  const float* wc = w + c * 11;
  float s = bias[c];
  if (vertical) {
#pragma unroll
    for (int t = 0; t < 11; ++t) {
      int yy = y + t - 5;
      if ((unsigned)yy < 64u) s += base[yy * 64 + x] * wc[t];
    }
  } else {
#pragma unroll
    for (int t = 0; t < 11; ++t) {
      int xx = x + t - 5;
      if ((unsigned)xx < 64u) s += base[y * 64 + xx] * wc[t];
    }
  }
  if (accumulate) out[idx] += s; else out[idx] = s;
}

// BN2 + ReLU -> qbf [B,C,N] bf16 (coalesced); qT produced by k_qtrans
__global__ void k_bn2q(const float* __restrict__ acc,
                       const float* __restrict__ bnA, const float* __restrict__ bnB,
                       __bf16* __restrict__ qbf) {
  int idx = blockIdx.x * 256 + threadIdx.x;
  if (idx >= BATCH * CR * NPIX) return;
  int c = (idx >> 12) & (CR - 1);
  float v = acc[idx] * bnA[c] + bnB[c];
  v = v > 0.f ? v : 0.f;
  qbf[idx] = (__bf16)v;
}

// qbf [B,C,N] -> qT [B,N,C] via LDS 32x32 bf16 tile transpose
__global__ __launch_bounds__(256) void k_qtrans(const __bf16* __restrict__ qbf,
                                                __bf16* __restrict__ qT) {
  __shared__ __bf16 t[32][33];
  int blk = blockIdx.x;                 // b * 8 * 128
  int pblk = blk & 127;
  int cblk = (blk >> 7) & 7;            // 256/32
  int b = blk >> 10;
  int p0 = pblk * 32, c0 = cblk * 32;
  int tp = threadIdx.x & 31, tc = threadIdx.x >> 5;
#pragma unroll
  for (int r = 0; r < 4; ++r) {
    int c = tc + r * 8;
    t[c][tp] = qbf[((size_t)(b * CR + c0 + c)) * NPIX + p0 + tp];
  }
  __syncthreads();
#pragma unroll
  for (int r = 0; r < 4; ++r) {
    int p = tc + r * 8;
    qT[((size_t)(b * NPIX + p0 + p)) * CR + c0 + tp] = t[tp][p];
  }
}

// column norms ||q[:,i]||
__global__ void k_norms(const __bf16* __restrict__ qbf,
                        float* __restrict__ rnorm) {
  int idx = blockIdx.x * 256 + threadIdx.x;     // B*N
  if (idx >= BATCH * NPIX) return;
  int i = idx & (NPIX - 1), b = idx >> 12;
  const __bf16* qc = qbf + (size_t)b * CR * NPIX + i;
  float s = 0.f;
  for (int c = 0; c < CR; ++c) {
    float v = (float)qc[(size_t)c * NPIX];
    s += v * v;
  }
  rnorm[idx] = sqrtf(s);
}

// Sv[b][c] = sum_j q[c,j]   (one block per (b,c))
__global__ void k_colsum(const __bf16* __restrict__ qbf,
                         float* __restrict__ Sv) {
  __shared__ float red[256];
  int bc = blockIdx.x;
  const __bf16* row = qbf + (size_t)bc * NPIX;
  float s = 0.f;
  for (int j = threadIdx.x; j < NPIX; j += 256) s += (float)row[j];
  red[threadIdx.x] = s;
  __syncthreads();
  for (int k = 128; k > 0; k >>= 1) {
    if (threadIdx.x < k) red[threadIdx.x] += red[threadIdx.x + k];
    __syncthreads();
  }
  if (threadIdx.x == 0) Sv[bc] = red[0];
}

// ---------------------------------------------------------------------------
// Fused cosine-similarity attention:
//   intra[i,c] = sum_j (q_i.q_j)/(|q_i||q_j|+eps) * q[c,j]
// One wave per 16-row tile, persistent 16x256 fp32 accumulators, streaming j.
// S tile is re-striped column->row distribution via a wave-private LDS tile.
// ---------------------------------------------------------------------------
__global__ __launch_bounds__(128) void k_attn(
    const __bf16* __restrict__ qT,   // [B,N,C]
    const __bf16* __restrict__ qbf,  // [B,C,N]
    const float* __restrict__ rnorm, // [B,N]
    float* __restrict__ intra) {     // [B,N,C] fp32
  __shared__ __attribute__((aligned(64))) __bf16 lds[4][16][32];
  const int lane = threadIdx.x & 31;
  const int wave = threadIdx.x >> 5;
  const int rt = blockIdx.x * 4 + wave;     // 0..B*256-1
  const int b = rt >> 8;
  const int i0 = (rt & 255) << 4;
  const __bf16* qTb = qT + (size_t)b * NPIX * CR;
  const __bf16* qb  = qbf + (size_t)b * CR * NPIX;
  const float* rn = rnorm + b * NPIX;
  const int h = lane >> 4, ncol = lane & 15;

  float ni[8];
#pragma unroll
  for (int r = 0; r < 8; ++r) ni[r] = rn[i0 + 8 * h + r];

  f32x8 acc[16];
#pragma unroll
  for (int ct = 0; ct < 16; ++ct) acc[ct] = zero8();

  const __bf16* arow = qTb + (size_t)(i0 + ncol) * CR;

  for (int j0 = 0; j0 < NPIX; j0 += 32) {
    f32x8 s0 = zero8(), s1 = zero8();
#pragma unroll
    for (int kt = 0; kt < 8; ++kt) {
      bf16x8 alo = *(const bf16x8*)(arow + kt * 32 + 16 * h);
      bf16x8 ahi = *(const bf16x8*)(arow + kt * 32 + 16 * h + 8);
      bf16x16 af = cat8(alo, ahi);
      bf16x16 b0 = *(const bf16x16*)(qTb + (size_t)(j0 + ncol) * CR + kt * 32 + 16 * h);
      bf16x16 b1 = *(const bf16x16*)(qTb + (size_t)(j0 + 16 + ncol) * CR + kt * 32 + 16 * h);
      s0 = wmma_bf16(af, b0, s0);
      s1 = wmma_bf16(af, b1, s1);
    }
    // cosine normalization, re-stripe S (col-per-lane) -> A-frag (row-per-lane)
    float nj0 = rn[j0 + ncol], nj1 = rn[j0 + 16 + ncol];
#pragma unroll
    for (int r = 0; r < 8; ++r) {
      float p0 = s0[r] / (ni[r] * nj0 + 1e-7f);
      float p1 = s1[r] / (ni[r] * nj1 + 1e-7f);
      lds[wave][8 * h + r][ncol]      = (__bf16)p0;
      lds[wave][8 * h + r][16 + ncol] = (__bf16)p1;
    }
    asm volatile("s_wait_dscnt 0" ::: "memory");
    bf16x8 plo = *(const bf16x8*)&lds[wave][ncol][16 * h];
    bf16x8 phi = *(const bf16x8*)&lds[wave][ncol][16 * h + 8];
    bf16x16 pf = cat8(plo, phi);
#pragma unroll
    for (int ct = 0; ct < 16; ++ct) {
      bf16x16 vf = *(const bf16x16*)(qb + (size_t)(ct * 16 + ncol) * NPIX + j0 + 16 * h);
      acc[ct] = wmma_bf16(pf, vf, acc[ct]);
    }
  }
#pragma unroll
  for (int ct = 0; ct < 16; ++ct)
#pragma unroll
    for (int r = 0; r < 8; ++r)
      intra[((size_t)(b * NPIX + i0 + 8 * h + r)) * CR + ct * 16 + ncol] = acc[ct][r];
}

// ---------------------------------------------------------------------------
// 1x1 conv GEMM: out[o,i] = W[o,:] . X[i,:], X = intra or (Sv - intra).
// Writes BN+ReLU'd bf16 into the NHWC concat buffer at channel offset chOff.
// Wave: 64 couts x 16 pixels.
// ---------------------------------------------------------------------------
__global__ __launch_bounds__(128) void k_conv1x1(
    const float* __restrict__ X,    // [B,N,256] fp32
    const float* __restrict__ Sv,   // [B,256]
    int mode,                       // 0: X, 1: Sv - X
    const __bf16* __restrict__ wfrag,
    const float* __restrict__ bnA, const float* __restrict__ bnB,
    __bf16* __restrict__ nhwc, int chOff) {
  const int lane = threadIdx.x & 31;
  const int wave = threadIdx.x >> 5;
  const int wid = blockIdx.x * 4 + wave;       // 0..4095
  const int b = wid >> 10;
  const int rem = wid & 1023;
  const int mg = rem >> 8;                     // group of 64 couts
  const int pt = rem & 255;
  const int p0 = pt * 16;
  const int h = lane >> 4, ncol = lane & 15;

  f32x8 acc[4];
#pragma unroll
  for (int t = 0; t < 4; ++t) acc[t] = zero8();

  const float* xrow = X + ((size_t)(b * NPIX + p0 + ncol)) * CR + 16 * h;
  const float* svb = Sv + b * CR;

#pragma unroll
  for (int kt = 0; kt < 8; ++kt) {
    bf16x16 bfrag;
    const float* xp = xrow + kt * 32;
#pragma unroll
    for (int j = 0; j < 16; ++j) {
      float v = xp[j];
      if (mode) v = svb[kt * 32 + 16 * h + j] - v;
      bfrag[j] = (__bf16)v;
    }
    const __bf16* wb = wfrag + ((size_t)kt * 16 + mg * 4) * 512 + lane * 16;
#pragma unroll
    for (int t = 0; t < 4; ++t) {
      bf16x16 a = *(const bf16x16*)(wb + (size_t)t * 512);
      acc[t] = wmma_bf16(a, bfrag, acc[t]);
    }
  }
  __bf16* orow = nhwc + ((size_t)(b * NPIX + p0 + ncol)) * CTOT + chOff;
#pragma unroll
  for (int t = 0; t < 4; ++t) {
    int ob = mg * 64 + t * 16 + 8 * h;
    bf16x8 pk;
#pragma unroll
    for (int r = 0; r < 8; ++r) {
      float v = acc[t][r] * bnA[ob + r] + bnB[ob + r];
      v = v > 0.f ? v : 0.f;
      pk[r] = (__bf16)v;
    }
    *(bf16x8*)(orow + ob) = pk;
  }
}

// ---------------------------------------------------------------------------
// Host orchestration
// ---------------------------------------------------------------------------
extern "C" void kernel_launch(void* const* d_in, const int* in_sizes, int n_in,
                              void* d_out, int out_size, void* d_ws, size_t ws_size,
                              hipStream_t stream) {
  (void)in_sizes; (void)n_in; (void)out_size; (void)ws_size;
  const float* feat    = (const float*)d_in[0];
  const float* agg_w1  = (const float*)d_in[1];
  const float* agg_b1  = (const float*)d_in[2];
  const float* bn1g = (const float*)d_in[3], *bn1b = (const float*)d_in[4];
  const float* bn1m = (const float*)d_in[5], *bn1v = (const float*)d_in[6];
  const float* fl1w = (const float*)d_in[7],  *fl1b = (const float*)d_in[8];
  const float* fl2w = (const float*)d_in[9],  *fl2b = (const float*)d_in[10];
  const float* fr1w = (const float*)d_in[11], *fr1b = (const float*)d_in[12];
  const float* fr2w = (const float*)d_in[13], *fr2b = (const float*)d_in[14];
  const float* bn2g = (const float*)d_in[15], *bn2b = (const float*)d_in[16];
  const float* bn2m = (const float*)d_in[17], *bn2v = (const float*)d_in[18];
  const float* wintra = (const float*)d_in[19], *bintra = (const float*)d_in[20];
  const float* bnig = (const float*)d_in[21], *bnib = (const float*)d_in[22];
  const float* bnim = (const float*)d_in[23], *bniv = (const float*)d_in[24];
  const float* winter = (const float*)d_in[25], *binter = (const float*)d_in[26];
  const float* bneg = (const float*)d_in[27], *bneb = (const float*)d_in[28];
  const float* bnem = (const float*)d_in[29], *bnev = (const float*)d_in[30];
  const float* botw = (const float*)d_in[31], *botb = (const float*)d_in[32];
  const float* bnog = (const float*)d_in[33], *bnob = (const float*)d_in[34];
  const float* bnom = (const float*)d_in[35], *bnov = (const float*)d_in[36];

  // workspace layout (all sub-buffers 256B aligned by construction)
  char* w = (char*)d_ws;
  __bf16* featNHWC = (__bf16*)w;                 w += (size_t)BATCH*NPIX*CTOT*2;      // 83.9 MB
  __bf16* w1frag   = (__bf16*)w;                 w += (size_t)9*CINF*CR*2;            // 9.4 MB
  __bf16* botfrag  = (__bf16*)w;                 w += (size_t)9*CTOT*CR*2;            // 11.8 MB
  __bf16* wifrag   = (__bf16*)w;                 w += (size_t)CR*CR*2;                // 128 KB
  __bf16* wrfrag   = (__bf16*)w;                 w += (size_t)CR*CR*2;
  float* x1        = (float*)w;                  w += (size_t)BATCH*CR*NPIX*4;        // 16.8 MB (aliased: intra)
  float* tmp       = (float*)w;                  w += (size_t)BATCH*CR*NPIX*4;
  float* accb      = (float*)w;                  w += (size_t)BATCH*CR*NPIX*4;
  __bf16* qbf      = (__bf16*)w;                 w += (size_t)BATCH*CR*NPIX*2;        // 8.4 MB
  __bf16* qT       = (__bf16*)w;                 w += (size_t)BATCH*NPIX*CR*2;
  float* rnorm     = (float*)w;                  w += (size_t)BATCH*NPIX*4;
  float* Sv        = (float*)w;                  w += (size_t)BATCH*CR*4;
  float* bn        = (float*)w;                  w += (size_t)10*256*4;
  __bf16* zpad     = (__bf16*)w;                 // CTOT bf16 zero page
  float* intra     = x1;                         // reuse: x1 dead after depthwise

  float* bn1A = bn, *bn1B = bn+256, *bn2A = bn+512, *bn2B = bn+768;
  float* bniA = bn+1024, *bniB = bn+1280, *bneA = bn+1536, *bneB = bn+1792;
  float* bnoA = bn+2048, *bnoB = bn+2304;

  // 1) zero page, input -> NHWC bf16, weight blobs, BN folding
  k_zero<<<(CTOT + 255) / 256, 256, 0, stream>>>(zpad, CTOT);
  k_tonhwc<<<BATCH * 64 * 128, 256, 0, stream>>>(feat, featNHWC);
  k_wfrag<<<(CR*CINF*9)/256, 256, 0, stream>>>(agg_w1, w1frag, CINF, 9, CR*CINF*9);
  k_wfrag<<<(CR*CTOT*9)/256, 256, 0, stream>>>(botw, botfrag, CTOT, 9, CR*CTOT*9);
  k_wfrag<<<(CR*CR)/256, 256, 0, stream>>>(wintra, wifrag, CR, 1, CR*CR);
  k_wfrag<<<(CR*CR)/256, 256, 0, stream>>>(winter, wrfrag, CR, 1, CR*CR);
  k_bnprep<<<1, 256, 0, stream>>>(bn1g, bn1b, bn1m, bn1v, agg_b1, bn1A, bn1B);
  k_bnprep<<<1, 256, 0, stream>>>(bn2g, bn2b, bn2m, bn2v, nullptr, bn2A, bn2B);
  k_bnprep<<<1, 256, 0, stream>>>(bnig, bnib, bnim, bniv, bintra, bniA, bniB);
  k_bnprep<<<1, 256, 0, stream>>>(bneg, bneb, bnem, bnev, binter, bneA, bneB);
  k_bnprep<<<1, 256, 0, stream>>>(bnog, bnob, bnom, bnov, botb, bnoA, bnoB);

  // 2) agg conv1 (3x3, 2048->256) + BN1 + ReLU  -> x1 (NCHW fp32)
  {
    dim3 grid(CR / 64, (BATCH * NPIX) / 64);
    k_conv3x3<<<grid, 128, 0, stream>>>(featNHWC, CINF, w1frag, zpad,
                                        bn1A, bn1B, x1);
  }

  // 3) depthwise separable chains: xl = V(H(x1)), xr = H(V(x1)); acc = xl+xr
  const int dwB = (BATCH*CR*NPIX)/256;
  k_dw<<<dwB, 256, 0, stream>>>(x1,  fl1w, fl1b, tmp,  0, 0);  // horizontal
  k_dw<<<dwB, 256, 0, stream>>>(tmp, fl2w, fl2b, accb, 1, 0);  // vertical (store)
  k_dw<<<dwB, 256, 0, stream>>>(x1,  fr1w, fr1b, tmp,  1, 0);  // vertical
  k_dw<<<dwB, 256, 0, stream>>>(tmp, fr2w, fr2b, accb, 0, 1);  // horizontal (+=)

  // 4) BN2 + ReLU -> q bf16 [C,N]; transpose -> [N,C]; norms; column sums
  k_bn2q<<<dwB, 256, 0, stream>>>(accb, bn2A, bn2B, qbf);
  k_qtrans<<<BATCH * 8 * 128, 256, 0, stream>>>(qbf, qT);
  k_norms<<<(BATCH*NPIX)/256, 256, 0, stream>>>(qbf, rnorm);
  k_colsum<<<BATCH*CR, 256, 0, stream>>>(qbf, Sv);

  // 5) fused cosine-attention -> intra [B,N,C] fp32 (reuses x1 buffer)
  k_attn<<<(BATCH*256)/4, 128, 0, stream>>>(qT, qbf, rnorm, intra);

  // 6) 1x1 convs + BN + ReLU -> concat channels [2048,2304) and [2304,2560)
  k_conv1x1<<<(BATCH*1024)/4, 128, 0, stream>>>(intra, Sv, 0, wifrag, bniA, bniB,
                                                featNHWC, CINF);
  k_conv1x1<<<(BATCH*1024)/4, 128, 0, stream>>>(intra, Sv, 1, wrfrag, bneA, bneB,
                                                featNHWC, CINF + CR);

  // 7) bottleneck 3x3 conv (2560->256) + BN + ReLU -> d_out (NCHW fp32)
  {
    dim3 grid(CR / 64, (BATCH * NPIX) / 64);
    k_conv3x3<<<grid, 128, 0, stream>>>(featNHWC, CTOT, botfrag, zpad,
                                        bnoA, bnoB, (float*)d_out);
  }
}
